// Attention_51977694216592
// MI455X (gfx1250) — matrix operand order repaired and checked by reading, not compiled, not run on hardware
//
#include <hip/hip_runtime.h>
#include <hip/hip_bf16.h>

// ---------------------------------------------------------------------------
// ViT global attention (B=8, H=W=32, DIM=768, 12 heads, head_dim=64), gfx1250.
// All GEMMs: v_wmma_f32_16x16x32_bf16 (bf16 in, f32 accumulate).
// Flash attention (online softmax), never materializing 96x1024x1024 scores.
// This round: K/V staging in the attention kernel goes through the Tensor
// Data Mover (tensor_load_to_lds + s_wait_tensorcnt), double-buffered so the
// DMA of chunk i+1 overlaps the WMMA/softmax work of chunk i.
// ---------------------------------------------------------------------------

typedef __attribute__((ext_vector_type(16))) __bf16 v16bf;
typedef __attribute__((ext_vector_type(8)))  __bf16 v8bf;
typedef __attribute__((ext_vector_type(4)))  __bf16 v4bf;
typedef __attribute__((ext_vector_type(8)))  float  v8f;
typedef __attribute__((ext_vector_type(4)))  float  v4f;
typedef __attribute__((ext_vector_type(4)))  unsigned int v4u;
typedef __attribute__((ext_vector_type(8)))  int    v8i;
typedef __attribute__((ext_vector_type(4)))  int    v4i;

#define NB   8
#define HW   1024
#define DIM  768
#define NH   12
#define HD   64
#define BH   (NB * NH)           // 96
#define NQKV (3 * DIM)           // 2304
#define PLANE ((size_t)BH * HW * HD)   // 6291456 elements per q/k/v plane

#if defined(__gfx1250__) && __has_builtin(__builtin_amdgcn_tensor_load_to_lds)
#define HAVE_TDM 1
#else
#define HAVE_TDM 0
#endif

__device__ __forceinline__ v8bf ld8(const __bf16* p) { return *(const v8bf*)p; }
__device__ __forceinline__ v16bf cat8(v8bf lo, v8bf hi) {
    return __builtin_shufflevector(lo, hi, 0, 1, 2, 3, 4, 5, 6, 7,
                                           8, 9, 10, 11, 12, 13, 14, 15);
}

// ---------------------------------------------------------------------------
// TDM: issue a 2D tile load (data_size = 2B elements) into LDS.
//   lds_off  : byte offset of destination in LDS
//   gsrc     : global address of tile start
//   tile_d0  : elements per row,  tile_d1 : rows
//   stride0  : row stride in elements
// D# group0/group1 packed per CDNA5 ISA 8.3/8.4; tensor dims set huge so the
// OOB clamp never engages (all reads are in-bounds by construction).
// ---------------------------------------------------------------------------
__device__ __forceinline__ void tdm_load_to_lds(unsigned lds_off, const void* gsrc,
                                                unsigned tile_d0, unsigned tile_d1,
                                                unsigned stride0) {
#if HAVE_TDM
    const unsigned long long ga = (unsigned long long)(size_t)gsrc;
    const unsigned td0 = 1u << 24, td1 = 1u << 24;
    v4u g0;
    g0[0] = 1u;                                     // count=1, user mode
    g0[1] = lds_off;                                // lds_addr (bytes)
    g0[2] = (unsigned)ga;                           // global_addr[31:0]
    g0[3] = (unsigned)((ga >> 32) & 0x01ffffffu)    // global_addr[56:32]
          | (2u << 30);                             // type = 2 ("image")
    v8i g1;
    g1[0] = (int)(1u << 16);                        // data_size=1 (2 bytes)
    g1[1] = (int)((td0 & 0xffffu) << 16);           // tensor_dim0[15:0]
    g1[2] = (int)((td0 >> 16) | ((td1 & 0xffffu) << 16));
    g1[3] = (int)((td1 >> 16) | (tile_d0 << 16));   // tile_dim0
    g1[4] = (int)(tile_d1 & 0xffffu);               // tile_dim1 (tile_dim2=0)
    g1[5] = (int)stride0;                           // tensor_dim0_stride[31:0]
    g1[6] = 0;
    g1[7] = 0;
    const v4i z4 = {0, 0, 0, 0};
#if __clang_major__ >= 23
    const v8i z8 = {0, 0, 0, 0, 0, 0, 0, 0};
    __builtin_amdgcn_tensor_load_to_lds(g0, g1, z4, z4, z8, 0);
#else
    __builtin_amdgcn_tensor_load_to_lds(g0, g1, z4, z4, 0);
#endif
#endif
}

// ---------------------------------------------------------------------------
// Prep 1: x (fp32) -> bf16, 4 elements/thread.
// ---------------------------------------------------------------------------
__global__ __launch_bounds__(256) void cvt_x(const float* __restrict__ x,
                                             __bf16* __restrict__ xbf) {
    const size_t i = ((size_t)blockIdx.x * 256 + threadIdx.x) * 4;
    const v4f v = *(const v4f*)(x + i);
    v4bf o;
    o[0] = (__bf16)v[0]; o[1] = (__bf16)v[1];
    o[2] = (__bf16)v[2]; o[3] = (__bf16)v[3];
    *(v4bf*)(xbf + i) = o;
}

// ---------------------------------------------------------------------------
// Prep 2: weight (KxN fp32, row-major) -> bf16 column-major (NxK).
// ---------------------------------------------------------------------------
__global__ __launch_bounds__(256) void transpose_w(const float* __restrict__ w,
                                                   __bf16* __restrict__ wt,
                                                   int K, int N) {
    const int idx = blockIdx.x * 256 + threadIdx.x;
    const int n = idx / K;
    const int k = idx - n * K;
    wt[(size_t)n * K + k] = (__bf16)w[(size_t)k * N + n];
}

// ---------------------------------------------------------------------------
// Kernel 1: QKV projection.  Xbf(8192x768) @ WqkvT + b.
// Q,K planes: [head][hw][hd];  V plane transposed: [head][hd][hw].
// ---------------------------------------------------------------------------
__global__ __launch_bounds__(256) void qkv_gemm(const __bf16* __restrict__ xbf,
                                                const __bf16* __restrict__ wT,
                                                const float* __restrict__ bias,
                                                __bf16* __restrict__ qkv) {
    const int lane = threadIdx.x & 31;
    const int wave = threadIdx.x >> 5;
    const int g    = lane >> 4;
    const int l    = lane & 15;

    const int NT   = NQKV / 16;                 // 144 column tiles
    const int tile = blockIdx.x * 8 + wave;
    const int mt   = tile / NT;
    const int nt   = tile - mt * NT;
    const int m0   = mt * 16;
    const int n0   = nt * 16;

    const __bf16* ap = xbf + (size_t)(m0 + l) * DIM;
    const __bf16* bp = wT + (size_t)(n0 + l) * DIM;

    v8f acc = {};
    for (int kk = 0; kk < DIM; kk += 32) {
        const v16bf a = cat8(ld8(ap + kk + 8 * g), ld8(ap + kk + 16 + 8 * g));
        const v16bf b = cat8(ld8(bp + kk + 8 * g), ld8(bp + kk + 16 + 8 * g));
        acc = __builtin_amdgcn_wmma_f32_16x16x32_bf16(false, a, false, b,
                                                      (short)0, acc, false, false);
    }

    const int   col   = n0 + l;
    const float bv    = bias[col];
    const int   which = col / DIM;              // uniform per wave
    const int   rem   = col - which * DIM;
    const int   head  = rem >> 6;               // uniform per wave
    const int   hdi   = rem & 63;
    __bf16* plane = qkv + (size_t)which * PLANE;

#pragma unroll
    for (int r = 0; r < 8; ++r) {
        const int row = m0 + r + 8 * g;
        const int bi  = row >> 10;
        const int hw  = row & (HW - 1);
        const float o = acc[r] + bv;
        if (which == 2)  // V transposed: [bh][hd][key]
            plane[((size_t)(bi * NH + head) * HD + hdi) * HW + hw] = (__bf16)o;
        else
            plane[((size_t)(bi * NH + head) * HW + hw) * HD + hdi] = (__bf16)o;
    }
}

// ---------------------------------------------------------------------------
// Kernel 2: decomposed relative-position bias (VALU fp32, tiny).
// ---------------------------------------------------------------------------
__global__ __launch_bounds__(256) void rel_bias(const __bf16* __restrict__ qbf,
                                                const float* __restrict__ rph,
                                                const float* __restrict__ rpw,
                                                float* __restrict__ relh,
                                                float* __restrict__ relw) {
    const int idx = blockIdx.x * 256 + threadIdx.x;
    const int k   = idx & 63;
    const int q   = (idx >> 6) & (HW - 1);
    const int bh  = idx >> 16;
    const int isw = k >> 5;
    const int kk  = k & 31;
    const int hq  = q >> 5;
    const int wq  = q & 31;

    const float* R = isw ? (rpw + (size_t)(wq - kk + 31) * HD)
                         : (rph + (size_t)(hq - kk + 31) * HD);
    const __bf16* qp = qbf + ((size_t)bh * HW + q) * HD;

    float s = 0.f;
#pragma unroll
    for (int c = 0; c < HD; c += 8) {
        const v8bf qv = ld8(qp + c);
        const v4f  r0 = *(const v4f*)(R + c);
        const v4f  r1 = *(const v4f*)(R + c + 4);
#pragma unroll
        for (int j = 0; j < 4; ++j) {
            s += (float)qv[j] * r0[j];
            s += (float)qv[4 + j] * r1[j];
        }
    }
    float* dst = isw ? relw : relh;
    dst[((size_t)bh * HW + q) * 32 + kk] = s;
}

// ---------------------------------------------------------------------------
// Kernel 3: fused flash attention.  One block = one head * 128 queries
// (8 waves x 16 rows).  Per 32-key chunk: K (32x64 bf16, contiguous 4KB) and
// V-transposed (64x32 bf16, row stride 1024 elems) are DMA'd into LDS by the
// Tensor Data Mover, double-buffered so chunk i+1's DMA overlaps chunk i's
// 8 WMMAs + softmax.  Fragments come out of LDS as 16B reads.
// ---------------------------------------------------------------------------
__global__ __launch_bounds__(256) void attn(const __bf16* __restrict__ qbf,
                                            const __bf16* __restrict__ kbf,
                                            const __bf16* __restrict__ vtbf,
                                            const float* __restrict__ relh,
                                            const float* __restrict__ relw,
                                            __bf16* __restrict__ obf) {
    __shared__ alignas(16) __bf16 sK[2 * 32 * HD];  // 2 x 4 KB  [key_local][c]
    __shared__ alignas(16) __bf16 sV[2 * HD * 32];  // 2 x 4 KB  [hd][key_local]

    const int lane = threadIdx.x & 31;
    const int wave = threadIdx.x >> 5;
    const int g    = lane >> 4;
    const int l    = lane & 15;

    const int bh = blockIdx.x >> 3;
    const int q0 = ((blockIdx.x & 7) * 8 + wave) * 16;

    const __bf16* Qh = qbf + (size_t)bh * HW * HD;
    const __bf16* Kh = kbf + (size_t)bh * HW * HD;
    const __bf16* Vt = vtbf + (size_t)bh * HD * HW;   // [hd][key]

    // Loop-invariant B-fragments of Q^T (head_dim 64 -> two 32-chunks).
    const __bf16* qp = Qh + (size_t)(q0 + l) * HD;
    const v16bf qB0 = cat8(ld8(qp + 16 * g), ld8(qp + 16 * g + 8));
    const v16bf qB1 = cat8(ld8(qp + 32 + 16 * g), ld8(qp + 32 + 16 * g + 8));

    const int myq = q0 + l;
    const float* bias_h = relh + ((size_t)bh * HW + myq) * 32;
    const float* bias_w = relw + ((size_t)bh * HW + myq) * 32;

    // rel_w bias index (key&31) is loop-invariant per lane: preload 16 values.
    float bw[16];
#pragma unroll
    for (int t = 0; t < 2; ++t)
#pragma unroll
        for (int r = 0; r < 8; ++r) bw[t * 8 + r] = bias_w[t * 16 + r + 8 * g];

    float m_i = -1e30f, l_i = 0.f;
    v8f accO[4] = {v8f{}, v8f{}, v8f{}, v8f{}};

#if HAVE_TDM
    const unsigned ldsK = (unsigned)(size_t)(void*)sK;
    const unsigned ldsV = (unsigned)(size_t)(void*)sV;
    if (wave == 0) {   // prologue: DMA chunk 0 into buffer 0
        tdm_load_to_lds(ldsK, Kh, 2048, 1, 2048);
        tdm_load_to_lds(ldsV, Vt, 32, 64, HW);
    }
#endif

    for (int ic = 0; ic < 32; ++ic) {
        const int k0 = ic * 32;
#if HAVE_TDM
        const int cur = ic & 1;
        if (wave == 0) __builtin_amdgcn_s_wait_tensorcnt(0);
        __syncthreads();   // chunk ic visible to all waves; prev compute done
        if (wave == 0 && ic < 31) {   // DMA chunk ic+1 while computing chunk ic
            const int nxt = cur ^ 1;
            tdm_load_to_lds(ldsK + nxt * 4096, Kh + (size_t)(k0 + 32) * HD,
                            2048, 1, 2048);
            tdm_load_to_lds(ldsV + nxt * 4096, Vt + (k0 + 32), 32, 64, HW);
        }
        const __bf16* kbuf = sK + cur * (32 * HD);
        const __bf16* vbuf = sV + cur * (HD * 32);
#else
        __syncthreads();
        {
            const int t = threadIdx.x;
            ((uint4*)sK)[t] = ((const uint4*)(Kh + (size_t)(k0 + (t >> 3)) * HD))[t & 7];
            ((uint4*)sV)[t] = ((const uint4*)(Vt + (size_t)(t >> 2) * HW + k0))[t & 3];
        }
        __syncthreads();
        const __bf16* kbuf = sK;
        const __bf16* vbuf = sV;
#endif

        // ---- two S^T 16x16 tiles: D = K_tile(16x64) @ Q^T(64x16) ----
        v8f st[2];
#pragma unroll
        for (int t = 0; t < 2; ++t) {
            const __bf16* kr = kbuf + (t * 16 + l) * HD;
            const v16bf ka0 = cat8(ld8(kr + 16 * g), ld8(kr + 16 * g + 8));
            const v16bf ka1 = cat8(ld8(kr + 32 + 16 * g), ld8(kr + 32 + 16 * g + 8));
            v8f s = {};
            s = __builtin_amdgcn_wmma_f32_16x16x32_bf16(false, ka0, false, qB0,
                                                        (short)0, s, false, false);
            s = __builtin_amdgcn_wmma_f32_16x16x32_bf16(false, ka1, false, qB1,
                                                        (short)0, s, false, false);
            st[t] = s;
        }

        // rel_h bias is constant across the whole 32-key chunk
        const float bhv = bias_h[ic];

        float p[16];
        float cmax = -1e30f;
#pragma unroll
        for (int j = 0; j < 16; ++j) {
            const float sv = st[j >> 3][j & 7] * 0.125f + bhv + bw[j];
            p[j] = sv;
            cmax = fmaxf(cmax, sv);
        }
        cmax = fmaxf(cmax, __shfl_xor(cmax, 16, 32));

        const float m_new = fmaxf(m_i, cmax);
        const float resc  = __expf(m_i - m_new);
        float psum = 0.f;
#pragma unroll
        for (int j = 0; j < 16; ++j) {
            p[j] = __expf(p[j] - m_new);
            psum += p[j];
        }
        psum += __shfl_xor(psum, 16, 32);
        l_i = l_i * resc + psum;
        m_i = m_new;

        // Rescale O accumulators (broadcast per-row factor into O layout)
        float rf[8];
#pragma unroll
        for (int r = 0; r < 8; ++r) rf[r] = __shfl(resc, r + 8 * g, 32);
#pragma unroll
        for (int hb = 0; hb < 4; ++hb)
#pragma unroll
            for (int r = 0; r < 8; ++r) accO[hb][r] *= rf[r];

        // P is already in A-fragment layout
        v16bf pa;
#pragma unroll
        for (int j = 0; j < 16; ++j) pa[j] = (__bf16)p[j];

        // O += P(16x32) @ V(32x64); V B-fragments contiguous in LDS
#pragma unroll
        for (int hb = 0; hb < 4; ++hb) {
            const __bf16* vr = vbuf + (hb * 16 + l) * 32;
            const v16bf vb = cat8(ld8(vr + 8 * g), ld8(vr + 16 + 8 * g));
            accO[hb] = __builtin_amdgcn_wmma_f32_16x16x32_bf16(false, pa, false, vb,
                                                               (short)0, accO[hb],
                                                               false, false);
        }

#if HAVE_TDM
        __syncthreads();   // don't let the TDM refill a buffer still being read
#endif
    }

    // Finalize: O[row] /= l[row]; store bf16 head-major [bh][q][hd]
    const float inv = 1.f / l_i;
    float rinv[8];
#pragma unroll
    for (int r = 0; r < 8; ++r) rinv[r] = __shfl(inv, r + 8 * g, 32);

    __bf16* op = obf + ((size_t)bh * HW + q0) * HD;
#pragma unroll
    for (int hb = 0; hb < 4; ++hb)
#pragma unroll
        for (int r = 0; r < 8; ++r)
            op[(size_t)(r + 8 * g) * HD + hb * 16 + l] = (__bf16)(accO[hb][r] * rinv[r]);
}

// ---------------------------------------------------------------------------
// Kernel 4: output projection through the head permute.
// out(8192x768) = A @ proj_w + proj_b, fp32 output.
// ---------------------------------------------------------------------------
__global__ __launch_bounds__(256) void proj_gemm(const __bf16* __restrict__ abf,
                                                 const __bf16* __restrict__ wT,
                                                 const float* __restrict__ bias,
                                                 float* __restrict__ out) {
    const int lane = threadIdx.x & 31;
    const int wave = threadIdx.x >> 5;
    const int g    = lane >> 4;
    const int l    = lane & 15;

    const int NT   = DIM / 16;                  // 48 column tiles
    const int tile = blockIdx.x * 8 + wave;
    const int mt   = tile / NT;
    const int nt   = tile - mt * NT;
    const int m0   = mt * 16;
    const int n0   = nt * 16;

    const int arow = m0 + l;
    const int bi   = arow >> 10;
    const int hw   = arow & (HW - 1);
    const __bf16* bp = wT + (size_t)(n0 + l) * DIM;

    v8f acc = {};
    for (int kk = 0; kk < DIM; kk += 32) {
        const int ka = kk + 8 * g;
        const int kb = kk + 16 + 8 * g;
        const __bf16* a0 =
            abf + ((size_t)(bi * NH + (ka >> 6)) * HW + hw) * HD + (ka & 63);
        const __bf16* a1 =
            abf + ((size_t)(bi * NH + (kb >> 6)) * HW + hw) * HD + (kb & 63);
        const v16bf a = cat8(ld8(a0), ld8(a1));
        const v16bf b = cat8(ld8(bp + kk + 8 * g), ld8(bp + kk + 16 + 8 * g));
        acc = __builtin_amdgcn_wmma_f32_16x16x32_bf16(false, a, false, b,
                                                      (short)0, acc, false, false);
    }

    const float bv = bias[n0 + l];
#pragma unroll
    for (int r = 0; r < 8; ++r)
        out[(size_t)(m0 + r + 8 * g) * DIM + n0 + l] = acc[r] + bv;
}

// ---------------------------------------------------------------------------
// Launcher.  Inputs (fp32): x, qkv_w, qkv_b, proj_w, proj_b, rel_pos_h,
// rel_pos_w.  Output fp32 (B,H,W,DIM).  Workspace (~93 MB):
//  [q|k|v bf16 37.75MB][relh 12.6MB][relw 12.6MB][attn-out bf16 12.6MB]
//  [xbf 12.6MB][wqkvT 3.54MB][wprojT 1.18MB]
// ---------------------------------------------------------------------------
extern "C" void kernel_launch(void* const* d_in, const int* in_sizes, int n_in,
                              void* d_out, int out_size, void* d_ws, size_t ws_size,
                              hipStream_t stream) {
    const float* x      = (const float*)d_in[0];
    const float* qkv_w  = (const float*)d_in[1];
    const float* qkv_b  = (const float*)d_in[2];
    const float* proj_w = (const float*)d_in[3];
    const float* proj_b = (const float*)d_in[4];
    const float* rph    = (const float*)d_in[5];
    const float* rpw    = (const float*)d_in[6];

    char* ws = (char*)d_ws;
    size_t off = 0;
    __bf16* qkvbf = (__bf16*)(ws + off); off += 3 * PLANE * sizeof(__bf16);
    float*  relh  = (float*)(ws + off);  off += (size_t)BH * HW * 32 * sizeof(float);
    float*  relw  = (float*)(ws + off);  off += (size_t)BH * HW * 32 * sizeof(float);
    __bf16* obf   = (__bf16*)(ws + off); off += PLANE * sizeof(__bf16);
    __bf16* xbf   = (__bf16*)(ws + off); off += (size_t)NB * HW * DIM * sizeof(__bf16);
    __bf16* wqkvT = (__bf16*)(ws + off); off += (size_t)DIM * NQKV * sizeof(__bf16);
    __bf16* wprojT= (__bf16*)(ws + off); off += (size_t)DIM * DIM * sizeof(__bf16);

    // --- prep: bf16 conversion + weight transposes (one-time, tiny) ---
    cvt_x<<<6144, 256, 0, stream>>>(x, xbf);
    transpose_w<<<6912, 256, 0, stream>>>(qkv_w, wqkvT, DIM, NQKV);
    transpose_w<<<2304, 256, 0, stream>>>(proj_w, wprojT, DIM, DIM);

    // --- pipeline ---
    qkv_gemm<<<9216, 256, 0, stream>>>(xbf, wqkvT, qkv_b, qkvbf);
    rel_bias<<<24576, 256, 0, stream>>>(qkvbf, rph, rpw, relh, relw);
    attn<<<768, 256, 0, stream>>>(qkvbf, qkvbf + PLANE, qkvbf + 2 * PLANE,
                                  relh, relw, obf);
    proj_gemm<<<3072, 256, 0, stream>>>(obf, wprojT, proj_b, (float*)d_out);
}
